// MultiExitClassifier_34978213659210
// MI455X (gfx1250) — compile-verified, matrix operand order checked by reading, not run on hardware
//
#include <hip/hip_runtime.h>
#include <math.h>

// ---------------- problem constants ----------------
#define BATCH   32768
#define HID     1024
#define NCLS    16
#define NLAYERS 12
#define LN_EPS  1e-5f

// ---------------- types ----------------
typedef __attribute__((ext_vector_type(16))) __bf16 v16bf;
typedef __attribute__((ext_vector_type(8)))  float  v8f;

union FragBF {
    v16bf v;
    uint4 u[2];
};

__device__ __forceinline__ unsigned short f2bf(float f) {
    unsigned u = __builtin_bit_cast(unsigned, f);
    u = (u + 0x7FFFu + ((u >> 16) & 1u)) >> 16;   // round-to-nearest-even
    return (unsigned short)u;
}

// Single-instruction packed f32->bf16 convert (RNE), replaces ~14 VALU ops.
__device__ __forceinline__ unsigned packbf(float a, float b) {
    unsigned r;
    asm("v_cvt_pk_bf16_f32 %0, %1, %2" : "=v"(r) : "v"(a), "v"(b));
    return r;
}

// CDNA5 async direct global->LDS copy (ASYNCcnt-tracked, no VGPR data).
// lds_off = wave-relative LDS byte offset (low 32 bits of flat shared addr).
__device__ __forceinline__ void async_load_b128(const void* gaddr, unsigned lds_off) {
    asm volatile("global_load_async_to_lds_b128 %0, %1, off"
                 :: "v"(lds_off), "v"(gaddr)
                 : "memory");
}
#define WAIT_ASYNC(n) asm volatile("s_wait_asynccnt %0" :: "i"(n) : "memory")

// ---------------- init: zero done flags + exit flags ----------------
__global__ void init_kernel(int* __restrict__ done, unsigned* __restrict__ flags) {
    int i = blockIdx.x * blockDim.x + threadIdx.x;
    if (i < BATCH) done[i] = 0;
    if (i < 4)     flags[i] = 0u;
}

// ---------------- one-time weight transpose + f32->bf16 -----------
// src: W[k][n] f32 row-major (z=0: W_in, z>=1: Ws layer z-1)
// dst: Wt[z][n][k] bf16
__global__ __launch_bounds__(256)
void transpose_convert_kernel(const float* __restrict__ W_in,
                              const float* __restrict__ Ws,
                              unsigned short* __restrict__ Wt) {
    __shared__ float tile[32][33];
    const int z = blockIdx.z;
    const float* src = (z == 0) ? W_in : (Ws + (size_t)(z - 1) * HID * HID);
    unsigned short* dst = Wt + (size_t)z * HID * HID;
    const int kBase = blockIdx.y * 32;
    const int nBase = blockIdx.x * 32;
    const int tx = threadIdx.x;          // 0..31
    const int ty = threadIdx.y;          // 0..7
#pragma unroll
    for (int i = 0; i < 32; i += 8)
        tile[ty + i][tx] = src[(size_t)(kBase + ty + i) * HID + nBase + tx];
    __syncthreads();
#pragma unroll
    for (int i = 0; i < 32; i += 8)
        dst[(size_t)(nBase + ty + i) * HID + kBase + tx] = f2bf(tile[tx][ty + i]);
}

// ---------------- bf16 WMMA GEMM: Y = act(A @ W + bias) -----------
// A: [BATCH][HID] f32, Wt: [HID(n)][HID(k)] bf16 (pre-transposed).
// Block computes 128(M) x 128(N); 8 waves, each wave 32x64 (2x4 wmma tiles).
// A staged via VGPRs (f32->bf16 cvt_pk); B staged via async DMA into a
// double-buffered LDS tile (pure bf16 copy, L2 -> LDS, no VGPR round-trip).
template <bool DO_GELU>
__global__ __launch_bounds__(256)
void gemm_kernel(const float* __restrict__ A,
                 const unsigned short* __restrict__ Wt,
                 const float* __restrict__ bias,
                 float* __restrict__ Y) {
    __shared__ unsigned short sA[128 * 40];      // 32 K-halves + 8 pad (80B row stride)
    __shared__ unsigned short sB[2][128 * 40];   // double-buffered async dest

    const int tid  = threadIdx.x;
    const int lane = tid & 31;
    const int wave = tid >> 5;
    const int wm   = wave >> 1;               // 0..3
    const int wn   = wave & 1;                // 0..1
    const int l15  = lane & 15;
    const int hi8  = (lane >> 4) << 3;        // 0 or 8
    const int ksel = hi8;                     // K-half select (halfwords)

    const int mBase = blockIdx.y * 128;
    const int nBase = blockIdx.x * 128;

    v8f acc[2][4];
#pragma unroll
    for (int tm = 0; tm < 2; ++tm)
#pragma unroll
        for (int tn = 0; tn < 4; ++tn) {
            v8f z = {0.f, 0.f, 0.f, 0.f, 0.f, 0.f, 0.f, 0.f};
            acc[tm][tn] = z;
        }

    // staging index maps
    const int arow = tid >> 3;                // 0..31  (+i*32)
    const int acol = (tid & 7) << 2;          // f32 quad within 32-wide chunk
    const int brow = tid >> 2;                // 0..63  (+i*64)
    const int bcol = (tid & 3) << 3;          // 8-halfword quad

    float4 ra[4];

    auto loadA = [&](int k0) {
#pragma unroll
        for (int i = 0; i < 4; ++i) {
            int r = arow + i * 32;
            ra[i] = *(const float4*)(A + (size_t)(mBase + r) * HID + k0 + acol);
        }
    };
    auto stageA = [&]() {
#pragma unroll
        for (int i = 0; i < 4; ++i) {
            int r = arow + i * 32;
            uint2 p;
            p.x = packbf(ra[i].x, ra[i].y);
            p.y = packbf(ra[i].z, ra[i].w);
            *(uint2*)(&sA[r * 40 + acol]) = p;
        }
    };
    auto issueB = [&](int buf, int k0) {
#pragma unroll
        for (int i = 0; i < 2; ++i) {
            int n = brow + i * 64;
            const unsigned short* g = Wt + (size_t)(nBase + n) * HID + k0 + bcol;
            unsigned loff = (unsigned)(size_t)(&sB[buf][n * 40 + bcol]);
            async_load_b128(g, loff);
        }
    };
    auto compute = [&](int buf) {
        FragBF a[2], b[4];
#pragma unroll
        for (int t = 0; t < 2; ++t) {
            int r = wm * 32 + t * 16 + l15;
            a[t].u[0] = *(const uint4*)(&sA[r * 40 + ksel]);
            a[t].u[1] = *(const uint4*)(&sA[r * 40 + ksel + 16]);
        }
#pragma unroll
        for (int t = 0; t < 4; ++t) {
            int n = wn * 64 + t * 16 + l15;
            b[t].u[0] = *(const uint4*)(&sB[buf][n * 40 + ksel]);
            b[t].u[1] = *(const uint4*)(&sB[buf][n * 40 + ksel + 16]);
        }
#pragma unroll
        for (int tm = 0; tm < 2; ++tm)
#pragma unroll
            for (int tn = 0; tn < 4; ++tn)
                acc[tm][tn] = __builtin_amdgcn_wmma_f32_16x16x32_bf16(
                    false, a[tm].v, false, b[tn].v,
                    (short)0, acc[tm][tn], false, false);
    };

    const int NK = HID / 32;
    issueB(0, 0);                         // async DMA chunk 0 into buf 0
    loadA(0);
    for (int kc = 0; kc < NK; ++kc) {
        __syncthreads();                  // prev sA consumed; sB[next] free (read 2 chunks ago)
        stageA();                         // regs -> sA (current chunk)
        if (kc + 1 < NK) {
            issueB((kc + 1) & 1, (kc + 1) * 32);   // DMA next B under this chunk's wmma
            loadA((kc + 1) * 32);                  // prefetch next A into regs
            WAIT_ASYNC(2);                // in-order: current-chunk B writes landed
        } else {
            WAIT_ASYNC(0);
        }
        __syncthreads();                  // all waves: sA stores + async B visible
        compute(kc & 1);
    }

    // epilogue: bias (+ exact GELU), store f32
#pragma unroll
    for (int tn = 0; tn < 4; ++tn) {
        int col = nBase + wn * 64 + tn * 16 + l15;
        float bv = bias[col];
#pragma unroll
        for (int tm = 0; tm < 2; ++tm) {
#pragma unroll
            for (int i = 0; i < 8; ++i) {
                int row = mBase + wm * 32 + tm * 16 + hi8 + i;
                float v = acc[tm][tn][i] + bv;
                if (DO_GELU)
                    v = 0.5f * v * (1.0f + erff(v * 0.70710678118654752f));
                Y[(size_t)row * HID + col] = v;
            }
        }
    }
}

// ---------------- LayerNorm: h[row] = LN(y[row])*gamma+beta if !done ---
__global__ __launch_bounds__(256)
void ln_kernel(const float* __restrict__ Y, float* __restrict__ Hout,
               const float* __restrict__ gamma, const float* __restrict__ beta,
               const int* __restrict__ done) {
    const int row = blockIdx.x;
    if (done[row]) return;               // done rows keep stale h (where(active))
    const int t = threadIdx.x;           // 256 threads, 4 floats each
    float4 v = *(const float4*)(Y + (size_t)row * HID + t * 4);
    float s  = v.x + v.y + v.z + v.w;
    float ss = v.x * v.x + v.y * v.y + v.z * v.z + v.w * v.w;
#pragma unroll
    for (int off = 16; off > 0; off >>= 1) {
        s  += __shfl_xor(s, off, 32);
        ss += __shfl_xor(ss, off, 32);
    }
    __shared__ float red[16];
    const int wv = t >> 5, lane = t & 31;
    if (lane == 0) { red[wv] = s; red[wv + 8] = ss; }
    __syncthreads();
    if (t == 0) {
        float ts = 0.f, tss = 0.f;
#pragma unroll
        for (int w = 0; w < 8; ++w) { ts += red[w]; tss += red[w + 8]; }
        red[0] = ts; red[8] = tss;
    }
    __syncthreads();
    float mu  = red[0] * (1.0f / HID);
    float var = red[8] * (1.0f / HID) - mu * mu;
    float rstd = rsqrtf(var + LN_EPS);
    float4 g = *(const float4*)(gamma + t * 4);
    float4 b = *(const float4*)(beta + t * 4);
    float4 o;
    o.x = (v.x - mu) * rstd * g.x + b.x;
    o.y = (v.y - mu) * rstd * g.y + b.y;
    o.z = (v.z - mu) * rstd * g.z + b.z;
    o.w = (v.w - mu) * rstd * g.w + b.w;
    *(float4*)(Hout + (size_t)row * HID + t * 4) = o;
}

// ---------------- exit head: logits, confidence, exit bookkeeping -----
__global__ __launch_bounds__(256)
void head_kernel(const float* __restrict__ H,
                 const float* __restrict__ Wh,   // [HID][NCLS] f32
                 const float* __restrict__ bh,   // [NCLS]
                 float* __restrict__ out,        // final_logits [BATCH][NCLS]
                 int* __restrict__ done,
                 unsigned* __restrict__ flags,
                 float thresh, int exit_idx, int is_last) {
    const int row = blockIdx.x * blockDim.x + threadIdx.x;
    if (row >= BATCH) return;
    if (done[row]) return;               // already exited: logits irrelevant

    float acc[NCLS];
#pragma unroll
    for (int c = 0; c < NCLS; ++c) acc[c] = bh[c];

    const float* hrow = H + (size_t)row * HID;
    for (int k = 0; k < HID; k += 4) {
        float4 hv = *(const float4*)(hrow + k);
#pragma unroll
        for (int j = 0; j < 4; ++j) {
            float hj = (j == 0) ? hv.x : (j == 1) ? hv.y : (j == 2) ? hv.z : hv.w;
            const float* wr = Wh + (size_t)(k + j) * NCLS;   // wave-uniform -> scalar loads
#pragma unroll
            for (int c = 0; c < NCLS; ++c) acc[c] += hj * wr[c];
        }
    }

    float m = acc[0];
#pragma unroll
    for (int c = 1; c < NCLS; ++c) m = fmaxf(m, acc[c]);
    float s = 0.f;
#pragma unroll
    for (int c = 0; c < NCLS; ++c) s += expf(acc[c] - m);
    float conf = 1.0f / s;               // max softmax prob = exp(0)/sum

    bool newly = conf >= thresh;
    if (newly || is_last) {
#pragma unroll
        for (int c = 0; c < NCLS; ++c) out[(size_t)row * NCLS + c] = acc[c];
    }
    if (newly) {
        done[row] = 1;
        flags[exit_idx] = 1u;            // benign race: all writers store 1
    }
}

// ---------------- finalize last_exit scalar --------------------------
__global__ void finalize_kernel(const unsigned* __restrict__ flags,
                                float* __restrict__ out_tail) {
    int le = flags[2] ? 12 : (flags[1] ? 6 : (flags[0] ? 3 : 12));
    out_tail[0] = (float)le;
}

// ---------------- host orchestration --------------------------------
extern "C" void kernel_launch(void* const* d_in, const int* in_sizes, int n_in,
                              void* d_out, int out_size, void* d_ws, size_t ws_size,
                              hipStream_t stream) {
    const float* x      = (const float*)d_in[0];   // (B, 1024)
    const float* W_in   = (const float*)d_in[1];   // (1024, 1024)
    const float* b_in   = (const float*)d_in[2];   // (1024,)
    const float* Ws     = (const float*)d_in[3];   // (12, 1024, 1024)
    const float* bs     = (const float*)d_in[4];   // (12, 1024)
    const float* gammas = (const float*)d_in[5];   // (12, 1024)
    const float* betas  = (const float*)d_in[6];   // (12, 1024)
    const float* Whs    = (const float*)d_in[7];   // (3, 1024, 16)
    const float* bhs    = (const float*)d_in[8];   // (3, 16)
    float* out = (float*)d_out;                    // [B*16 logits][1 last_exit]

    // workspace layout (~296 MB): h | y | Wt(bf16 x13) | done | flags
    float* h = (float*)d_ws;
    float* y = h + (size_t)BATCH * HID;
    unsigned short* Wt = (unsigned short*)(y + (size_t)BATCH * HID);
    int* done = (int*)(Wt + (size_t)13 * HID * HID);
    unsigned* flags = (unsigned*)(done + BATCH);

    const dim3 gemmGrid(HID / 128, BATCH / 128);   // (8, 256)

    init_kernel<<<(BATCH + 255) / 256, 256, 0, stream>>>(done, flags);
    transpose_convert_kernel<<<dim3(HID / 32, HID / 32, 13), dim3(32, 8), 0, stream>>>(
        W_in, Ws, Wt);

    // h = x @ W_in + b_in (no activation, no LN)
    gemm_kernel<false><<<gemmGrid, 256, 0, stream>>>(x, Wt, b_in, h);

    const float thr[3] = {0.95f, 0.92f, 0.88f};
    int exit_idx = 0;
    for (int l = 0; l < NLAYERS; ++l) {
        // y = GELU(h @ Ws[l] + bs[l])  (computed for all rows, like reference)
        gemm_kernel<true><<<gemmGrid, 256, 0, stream>>>(
            h, Wt + (size_t)(1 + l) * HID * HID, bs + (size_t)l * HID, y);
        // h = LN(y)*gamma+beta for active rows only
        ln_kernel<<<BATCH, 256, 0, stream>>>(
            y, h, gammas + (size_t)l * HID, betas + (size_t)l * HID, done);
        if (l == 2 || l == 5 || l == 11) {
            head_kernel<<<BATCH / 256, 256, 0, stream>>>(
                h, Whs + (size_t)exit_idx * HID * NCLS, bhs + (size_t)exit_idx * NCLS,
                out, done, flags, thr[exit_idx], exit_idx, (exit_idx == 2) ? 1 : 0);
            ++exit_idx;
        }
    }
    finalize_kernel<<<1, 1, 0, stream>>>(flags, out + (size_t)BATCH * NCLS);
}
